// Encoder_75557064671564
// MI455X (gfx1250) — matrix operand order repaired and checked by reading, not compiled, not run on hardware
//
#include <hip/hip_runtime.h>
#include <stdint.h>

// ---------------------------------------------------------------- types ----
typedef __bf16 bf16;
typedef __attribute__((ext_vector_type(16))) __bf16 v16bf;
typedef __attribute__((ext_vector_type(8)))  float  v8f;

#define BATCH 8
#define HH 256
#define WW 256
#define HW 65536

__device__ __forceinline__ float bf2f(bf16 v) {
    unsigned short u = __builtin_bit_cast(unsigned short, v);
    unsigned int x = ((unsigned int)u) << 16;
    return __builtin_bit_cast(float, x);
}
__device__ __forceinline__ bf16 f2bf(float f) {
    unsigned int x = __builtin_bit_cast(unsigned int, f);
    unsigned int lsb = (x >> 16) & 1u;
    x += 0x7FFFu + lsb;                       // round to nearest even
    unsigned short u = (unsigned short)(x >> 16);
    return __builtin_bit_cast(bf16, u);
}

// ---------------------------------------------------------------- utils ----
__global__ void fill_u32(uint32_t* p, uint32_t v, long long n) {
    long long i = (long long)blockIdx.x * blockDim.x + threadIdx.x;
    long long st = (long long)gridDim.x * blockDim.x;
    for (; i < n; i += st) p[i] = v;
}

// src: OIHW f32 [64][I][3][3]  ->  dst bf16 [m][kpos][Cp] (in-ch zero padded)
__global__ void cvt_weights(const float* __restrict__ src, bf16* __restrict__ dst,
                            int I, int Cp) {
    int n = 64 * 9 * Cp;
    for (int i = blockIdx.x * blockDim.x + threadIdx.x; i < n;
         i += gridDim.x * blockDim.x) {
        int c  = i % Cp;
        int kp = (i / Cp) % 9;
        int m  = i / (9 * Cp);
        float v = (c < I) ? src[(m * I + c) * 9 + kp] : 0.f;
        dst[i] = f2bf(v);
    }
}

// ---------------------------------------------------------------- FFT ------
// 256-point in-LDS radix-2 DIT FFT, 128 threads, sign = -1 fwd / +1 inv.
__device__ __forceinline__ void fft256_lds(float2* s, int tid, float sign) {
    for (int e = tid; e < 256; e += 128) {
        int r = __brev(e) >> 24;
        if (r > e) { float2 t = s[e]; s[e] = s[r]; s[r] = t; }
    }
    __syncthreads();
    for (int len = 2; len <= 256; len <<= 1) {
        int half = len >> 1;
        int grp = tid / half;
        int k   = tid - grp * half;
        int p   = grp * len + k;
        float ang = sign * 6.28318530717958647692f * (float)k / (float)len;
        float sn, cs;
        __sincosf(ang, &sn, &cs);
        float2 u = s[p], v = s[p + half];
        float2 w = make_float2(v.x * cs - v.y * sn, v.x * sn + v.y * cs);
        s[p]        = make_float2(u.x + w.x, u.y + w.y);
        s[p + half] = make_float2(u.x - w.x, u.y - w.y);
        __syncthreads();
    }
}

// forward row FFT of real planes: src f32 [p][256][256] -> dst complex
__global__ void fft_rows_f2c(const float* __restrict__ src, float2* __restrict__ dst) {
    __shared__ float2 s[256];
    int p = blockIdx.y, row = blockIdx.x, tid = threadIdx.x;
    const float* rp = src + (size_t)p * HW + row * 256;
    for (int e = tid; e < 256; e += 128) s[e] = make_float2(rp[e], 0.f);
    __syncthreads();
    fft256_lds(s, tid, -1.f);
    float2* op = dst + (size_t)p * HW + row * 256;
    for (int e = tid; e < 256; e += 128) op[e] = s[e];
}

// column FFT (either direction); optional low-pass mask (radius 50) after fwd
__global__ void fft_cols(const float2* __restrict__ src, float2* __restrict__ dst,
                         float sign, float scale, int mask) {
    __shared__ float2 s[256];
    int p = blockIdx.y, col = blockIdx.x, tid = threadIdx.x;
    const float2* cp = src + (size_t)p * HW + col;
    for (int e = tid; e < 256; e += 128) s[e] = cp[(size_t)e * 256];
    __syncthreads();
    fft256_lds(s, tid, sign);
    float2* op = dst + (size_t)p * HW + col;
    int fx = (col < 128) ? col : col - 256;
    for (int e = tid; e < 256; e += 128) {
        float2 v = s[e];
        v.x *= scale; v.y *= scale;
        if (mask) {
            int fy = (e < 128) ? e : e - 256;
            if (fy * fy + fx * fx > 2500) { v.x = 0.f; v.y = 0.f; }
        }
        op[(size_t)e * 256] = v;
    }
}

// inverse row FFT + postprocess.
// mode 0: low-pass -> lpb bf16 [b][32ch pad][256][256], floor(clip)/255
// mode 1: watermark -> concat channel 30 (enc ch0), bf16
__global__ void fft_rows_inv_post(const float2* __restrict__ src,
                                  bf16* __restrict__ out, int mode) {
    __shared__ float2 s[256];
    int p = blockIdx.y, row = blockIdx.x, tid = threadIdx.x;
    const float2* rp = src + (size_t)p * HW + row * 256;
    for (int e = tid; e < 256; e += 128) s[e] = rp[e];
    __syncthreads();
    fft256_lds(s, tid, 1.f);
    for (int e = tid; e < 256; e += 128) {
        float re = s[e].x * (1.f / 256.f);
        if (mode == 0) {
            int b = p / 3, c = p % 3;
            float v = floorf(fminf(fmaxf(re, 0.f), 255.f)) * (1.f / 255.f);
            out[(((size_t)(b * 32 + c)) << 16) + row * 256 + e] = f2bf(v);
        } else {
            out[(((size_t)(p * 128 + 30)) << 16) + row * 256 + e] = f2bf(re);
        }
    }
}

// 30 fixed tree-ring positions (all channel 0), rows/cols around center 128
__constant__ int WMR[30] = {124, 125,125,125,125,125, 126,126,126,126,126,126,126,
                            127,127,127,127,127,127,127,
                            128,128,128,128,128,128,128,128,128, 129};
__constant__ int WMC[30] = {128, 126,127,128,129,130, 125,126,127,128,129,130,131,
                            125,126,127,128,129,130,131,
                            124,125,126,127,128,129,130,131,132, 125};

__global__ void wm_scatter(float2* __restrict__ F, const float* __restrict__ msg) {
    int t = threadIdx.x;
    if (t < BATCH * 30) {
        int b = t / 30, l = t % 30;
        float m = msg[b * 30 + l];
        F[(size_t)b * HW + WMR[l] * 256 + WMC[l]] = make_float2(m, m);  // m*(1+1j)
    }
}

// ------------------------------------------------------- conv3x3 (WMMA) ----
// Implicit GEMM: M=64 out-ch, N=64 pixels (row segment), K=Cp*9.
// in  : bf16 [B][Cp][256][256]   (Cp = 32/64/128, padded channels are zero)
// wt  : bf16 [m][kpos][Cp]
// y   : f32  [B][64][256][256]   (pre-BN)
__launch_bounds__(256)
__global__ void conv3x3_wmma(const bf16* __restrict__ in, const bf16* __restrict__ wt,
                             float* __restrict__ y, int Cp) {
    __shared__ bf16 wlds[9 * 64 * 32];   // [kpos][m][k]   36 KB
    __shared__ bf16 ilds[3 * 66 * 32];   // [r][col][ch]   12.4 KB (ch contiguous)

    const int t    = blockIdx.x;
    const int x0   = (t & 3) << 6;
    const int row  = (t >> 2) & 255;
    const int b    = t >> 10;
    const int tid  = threadIdx.x;
    const int lane = tid & 31;
    const int wave = tid >> 5;
    const int mt   = wave & 3;            // 16-row out-channel tile
    const int ntb  = (wave >> 2) << 1;    // two 16-pixel tiles per wave

    v8f acc0 = {}; v8f acc1 = {};
    const int icgN = Cp >> 5;

    for (int icg = 0; icg < icgN; ++icg) {
        __syncthreads();
        // stage weight slice: 9*64*32 bf16
        for (int i = tid; i < 9 * 64 * 32; i += 256) {
            int k  = i & 31;
            int m  = (i >> 5) & 63;
            int kp = i >> 11;
            wlds[i] = wt[(m * 9 + kp) * Cp + (icg << 5) + k];
        }
        // stage haloed input patch: 3 rows x 66 cols x 32 ch (transposed, zero pad)
        for (int i = tid; i < 3 * 66 * 32; i += 256) {
            int ch  = i & 31;
            int cr  = i >> 5;
            int col = cr % 66;
            int r   = cr / 66;
            int gx  = x0 - 1 + col;
            int gy  = row - 1 + r;
            bf16 v = f2bf(0.f);
            if ((unsigned)gx < 256u && (unsigned)gy < 256u)
                v = in[((b * Cp + (icg << 5) + ch) * 256 + gy) * 256 + gx];
            ilds[i] = v;
        }
        __syncthreads();

#pragma unroll
        for (int kp = 0; kp < 9; ++kp) {
            const int ky = kp / 3, kx = kp % 3;
            union { v16bf v; uint32_t u[8]; } A, B0, B1;
            // A fragment (contiguous k = (lane/16)*16 + s): m = mt*16 + lane%16
            const uint32_t* wp = (const uint32_t*)
                (wlds + kp * 2048 + ((mt << 4) + (lane & 15)) * 32);
#pragma unroll
            for (int q = 0; q < 8; ++q)
                A.u[q] = wp[((lane >> 4) << 3) + q];
            // B fragments: n = nt*16 + lane%16 ; K pairs (lane/16)*16 + 2q
            const int nlo  = lane & 15;
            const int kseg = (lane >> 4) << 3;   // dword offset
            const uint32_t* ip0 = (const uint32_t*)
                (ilds + (ky * 66 + (ntb << 4) + nlo + kx) * 32);
            const uint32_t* ip1 = (const uint32_t*)
                (ilds + (ky * 66 + ((ntb + 1) << 4) + nlo + kx) * 32);
#pragma unroll
            for (int q = 0; q < 8; ++q) { B0.u[q] = ip0[kseg + q]; B1.u[q] = ip1[kseg + q]; }

            acc0 = __builtin_amdgcn_wmma_f32_16x16x32_bf16(
                false, A.v, false, B0.v, (short)0, acc0, false, false);
            acc1 = __builtin_amdgcn_wmma_f32_16x16x32_bf16(
                false, A.v, false, B1.v, (short)0, acc1, false, false);
        }
    }
    // C layout: VGPR r -> m = r + 8*(lane/16); n = lane%16
#pragma unroll
    for (int r = 0; r < 8; ++r) {
        int m = (mt << 4) + ((lane >> 4) << 3) + r;
        float* yp = y + (((b << 6) + m) * 256 + row) * 256 + x0;
        yp[(ntb << 4) + (lane & 15)]       = acc0[r];
        yp[((ntb + 1) << 4) + (lane & 15)] = acc1[r];
    }
}

// --------------------------------------------------------------- BN --------
// stats[c] = sum, stats[64+c] = sumsq  over B*H*W (atomic accumulation)
__global__ void bn_stats(const float* __restrict__ y, float* __restrict__ stats) {
    __shared__ float ss[256], sq[256];
    int c = blockIdx.x;
    int base = blockIdx.y * 8192;
    float s = 0.f, q = 0.f;
    for (int i = threadIdx.x; i < 8192; i += 256) {
        int e = base + i;
        int b = e >> 16, pix = e & 65535;
        float v = y[(((b << 6) + c) << 16) + pix];
        s += v; q += v * v;
    }
    ss[threadIdx.x] = s; sq[threadIdx.x] = q;
    __syncthreads();
    for (int o = 128; o > 0; o >>= 1) {
        if (threadIdx.x < o) {
            ss[threadIdx.x] += ss[threadIdx.x + o];
            sq[threadIdx.x] += sq[threadIdx.x + o];
        }
        __syncthreads();
    }
    if (threadIdx.x == 0) {
        atomicAdd(&stats[c], ss[0]);
        atomicAdd(&stats[64 + c], sq[0]);
    }
}

// normalize + ReLU -> bf16 into out (channel stride ocs, offset ooff);
// optionally mirror channel 0 in f32 to xc0 (FFT input)
__global__ void bn_apply(const float* __restrict__ y, const float* __restrict__ stats,
                         const float* __restrict__ g, const float* __restrict__ be,
                         bf16* __restrict__ out, int ocs, int ooff,
                         float* __restrict__ xc0) {
    const float invN = 1.f / 524288.f;
    long long n = (long long)BATCH * 64 * HW;
    for (long long e = (long long)blockIdx.x * blockDim.x + threadIdx.x; e < n;
         e += (long long)gridDim.x * blockDim.x) {
        int pix = (int)(e & 65535);
        int c   = (int)((e >> 16) & 63);
        int b   = (int)(e >> 22);
        float mean = stats[c] * invN;
        float var  = stats[64 + c] * invN - mean * mean;
        float sc   = g[c] * rsqrtf(var + 1e-5f);
        float v    = (y[e] - mean) * sc + be[c];
        v = fmaxf(v, 0.f);
        out[(((size_t)(b * ocs + ooff + c)) << 16) + pix] = f2bf(v);
        if (xc0 && c == 0) xc0[(((size_t)b) << 16) + pix] = v;
    }
}

// concat channels 0..29 = msg broadcast; 94..96 = image; 97..127 = 0
__global__ void fill_concat(const float* __restrict__ msg, const float* __restrict__ img,
                            bf16* __restrict__ out) {
    long long n = (long long)BATCH * 64 * HW;
    for (long long e = (long long)blockIdx.x * blockDim.x + threadIdx.x; e < n;
         e += (long long)gridDim.x * blockDim.x) {
        int pix = (int)(e & 65535);
        int c   = (int)((e >> 16) & 63);
        int b   = (int)(e >> 22);
        if (c < 30) {
            out[(((size_t)(b * 128 + c)) << 16) + pix] = f2bf(msg[b * 30 + c]);
        } else if (c < 33) {
            out[(((size_t)(b * 128 + 94 + (c - 30))) << 16) + pix] =
                f2bf(img[(((size_t)(b * 3 + (c - 30))) << 16) + pix]);
        } else {
            out[(((size_t)(b * 128 + 97 + (c - 33))) << 16) + pix] = f2bf(0.f);
        }
    }
}

// final 1x1 conv 64 -> 3 with bias, f32 out
__global__ void final_conv(const bf16* __restrict__ act, const float* __restrict__ fw,
                           const float* __restrict__ fb, float* __restrict__ out) {
    int n = BATCH * HW;
    for (int e = blockIdx.x * blockDim.x + threadIdx.x; e < n;
         e += gridDim.x * blockDim.x) {
        int b = e >> 16, pix = e & 65535;
        float a0 = fb[0], a1 = fb[1], a2 = fb[2];
        for (int c = 0; c < 64; ++c) {
            float v = bf2f(act[(((size_t)((b << 6) + c)) << 16) + pix]);
            a0 += fw[c] * v;
            a1 += fw[64 + c] * v;
            a2 += fw[128 + c] * v;
        }
        out[(((size_t)(b * 3)) << 16) + pix]     = a0;
        out[(((size_t)(b * 3 + 1)) << 16) + pix] = a1;
        out[(((size_t)(b * 3 + 2)) << 16) + pix] = a2;
    }
}

// --------------------------------------------------------------- driver ----
extern "C" void kernel_launch(void* const* d_in, const int* in_sizes, int n_in,
                              void* d_out, int out_size, void* d_ws, size_t ws_size,
                              hipStream_t stream) {
    const float* image   = (const float*)d_in[0];
    const float* message = (const float*)d_in[1];
    const float* w0  = (const float*)d_in[2];
    const float* g0  = (const float*)d_in[4];
    const float* be0 = (const float*)d_in[5];
    const float* wsw = (const float*)d_in[6];
    const float* gsw = (const float*)d_in[8];
    const float* besw= (const float*)d_in[9];
    const float* acw = (const float*)d_in[10];
    const float* acg = (const float*)d_in[12];
    const float* acbe= (const float*)d_in[13];
    const float* fw  = (const float*)d_in[14];
    const float* fb  = (const float*)d_in[15];

    uint8_t* base = (uint8_t*)d_ws;
    size_t off = 0;
    auto carve = [&](size_t bytes) -> void* {
        void* p = base + off;
        off = (off + bytes + 255) & ~(size_t)255;
        return p;
    };
    bf16*  lpb  = (bf16*)carve((size_t)BATCH * 32 * HW * 2);   // padded lp, bf16
    bf16*  actA = (bf16*)carve((size_t)BATCH * 64 * HW * 2);
    bf16*  actB = (bf16*)carve((size_t)BATCH * 64 * HW * 2);
    float* ybuf = (float*)carve((size_t)BATCH * 64 * HW * 4);  // pre-BN f32
    bf16*  cc   = (bf16*)carve((size_t)BATCH * 128 * HW * 2);  // concat (padded)
    float* xc0  = (float*)carve((size_t)BATCH * HW * 4);       // x channel-0 f32
    bf16*  wb0  = (bf16*)carve((size_t)64 * 9 * 32 * 2);
    bf16*  wbs0 = (bf16*)carve((size_t)64 * 9 * 64 * 2);
    bf16*  wbs1 = (bf16*)carve((size_t)64 * 9 * 64 * 2);
    bf16*  wbs2 = (bf16*)carve((size_t)64 * 9 * 64 * 2);
    bf16*  wbac = (bf16*)carve((size_t)64 * 9 * 128 * 2);
    float* stats= (float*)carve(512);
    // FFT scratch aliases ybuf (ybuf is idle whenever FFTs run, stream-ordered)
    float2* fA = (float2*)ybuf;
    float2* fB = fA + (size_t)24 * HW;

    // zero padded lp buffer, convert/pad weights to bf16 [m][kpos][Cp]
    fill_u32<<<4096, 256, 0, stream>>>((uint32_t*)lpb, 0u,
                                       (long long)BATCH * 32 * HW * 2 / 4);
    cvt_weights<<<64, 256, 0, stream>>>(w0, wb0, 3, 32);
    cvt_weights<<<64, 256, 0, stream>>>(wsw + 0 * 64 * 64 * 9, wbs0, 64, 64);
    cvt_weights<<<64, 256, 0, stream>>>(wsw + 1 * 64 * 64 * 9, wbs1, 64, 64);
    cvt_weights<<<64, 256, 0, stream>>>(wsw + 2 * 64 * 64 * 9, wbs2, 64, 64);
    cvt_weights<<<128, 256, 0, stream>>>(acw, wbac, 97, 128);

    // low-pass filter: 24 planes, fwd rows -> fwd cols + mask -> inv cols -> inv rows
    fft_rows_f2c<<<dim3(256, 24), 128, 0, stream>>>(image, fA);
    fft_cols<<<dim3(256, 24), 128, 0, stream>>>(fA, fB, -1.f, 1.f, 1);
    fft_cols<<<dim3(256, 24), 128, 0, stream>>>(fB, fA, 1.f, 1.f / 256.f, 0);
    fft_rows_inv_post<<<dim3(256, 24), 128, 0, stream>>>(fA, lpb, 0);

    // conv0: 3(->32 pad) -> 64
    conv3x3_wmma<<<8192, 256, 0, stream>>>(lpb, wb0, ybuf, 32);
    fill_u32<<<1, 128, 0, stream>>>((uint32_t*)stats, 0u, 128);
    bn_stats<<<dim3(64, 64), 256, 0, stream>>>(ybuf, stats);
    bn_apply<<<8192, 256, 0, stream>>>(ybuf, stats, g0, be0, actA, 64, 0, nullptr);

    // conv1..2: 64 -> 64 (ping-pong)
    conv3x3_wmma<<<8192, 256, 0, stream>>>(actA, wbs0, ybuf, 64);
    fill_u32<<<1, 128, 0, stream>>>((uint32_t*)stats, 0u, 128);
    bn_stats<<<dim3(64, 64), 256, 0, stream>>>(ybuf, stats);
    bn_apply<<<8192, 256, 0, stream>>>(ybuf, stats, gsw, besw, actB, 64, 0, nullptr);

    conv3x3_wmma<<<8192, 256, 0, stream>>>(actB, wbs1, ybuf, 64);
    fill_u32<<<1, 128, 0, stream>>>((uint32_t*)stats, 0u, 128);
    bn_stats<<<dim3(64, 64), 256, 0, stream>>>(ybuf, stats);
    bn_apply<<<8192, 256, 0, stream>>>(ybuf, stats, gsw + 64, besw + 64, actA, 64, 0, nullptr);

    // conv3: output x written straight into concat channels 30..93 (+ ch0 f32 copy)
    conv3x3_wmma<<<8192, 256, 0, stream>>>(actA, wbs2, ybuf, 64);
    fill_u32<<<1, 128, 0, stream>>>((uint32_t*)stats, 0u, 128);
    bn_stats<<<dim3(64, 64), 256, 0, stream>>>(ybuf, stats);
    bn_apply<<<8192, 256, 0, stream>>>(ybuf, stats, gsw + 128, besw + 128, cc, 128, 30, xc0);

    // fill concat msg/image/pad regions
    fill_concat<<<8192, 256, 0, stream>>>(message, image, cc);

    // tree-ring watermark: only channel 0 per batch is touched -> 8 planes
    fft_rows_f2c<<<dim3(256, 8), 128, 0, stream>>>(xc0, fA);
    fft_cols<<<dim3(256, 8), 128, 0, stream>>>(fA, fB, -1.f, 1.f, 0);
    wm_scatter<<<1, 256, 0, stream>>>(fB, message);
    fft_cols<<<dim3(256, 8), 128, 0, stream>>>(fB, fA, 1.f, 1.f / 256.f, 0);
    fft_rows_inv_post<<<dim3(256, 8), 128, 0, stream>>>(fA, cc, 1);

    // after-concat conv: 97(->128 pad) -> 64
    conv3x3_wmma<<<8192, 256, 0, stream>>>(cc, wbac, ybuf, 128);
    fill_u32<<<1, 128, 0, stream>>>((uint32_t*)stats, 0u, 128);
    bn_stats<<<dim3(64, 64), 256, 0, stream>>>(ybuf, stats);
    bn_apply<<<8192, 256, 0, stream>>>(ybuf, stats, acg, acbe, actA, 64, 0, nullptr);

    // final 1x1 conv -> d_out f32 [8][3][256][256]
    final_conv<<<2048, 256, 0, stream>>>(actA, fw, fb, (float*)d_out);
}
